// HeteroGNN_83562883711210
// MI455X (gfx1250) — compile-verified, matrix operand order checked by reading, not run on hardware
//
#include <hip/hip_runtime.h>
#include <hip/hip_bf16.h>
#include <cstdint>
#include <cstddef>

typedef __bf16 bf16;
typedef __attribute__((ext_vector_type(16))) __bf16 bf16x16;
typedef __attribute__((ext_vector_type(8)))  float  f32x8;
typedef __attribute__((ext_vector_type(4)))  int    i32x4;

#define HH     512   // H*C
#define CH     128   // hidden per head
#define NHEAD  4
#define NB     64    // batch (num graphs)
#define NINNER 64
#define CEILDIV(a,b) (((a)+(b)-1)/(b))

#define AS1 __attribute__((address_space(1)))
#define AS3 __attribute__((address_space(3)))

#if __has_builtin(__builtin_amdgcn_global_load_async_to_lds_b128)
#define HAVE_ASYNC_LDS 1
#else
#define HAVE_ASYNC_LDS 0
#endif

// ---------------------------------------------------------------------------
// small helpers
// ---------------------------------------------------------------------------
__device__ __forceinline__ void atomicMaxF(float* addr, float v) {
  unsigned int* ua = (unsigned int*)addr;
  unsigned int old = *ua;
  while (__uint_as_float(old) < v) {
    unsigned int assumed = old;
    old = atomicCAS(ua, assumed, __float_as_uint(v));
    if (old == assumed) break;
  }
}

// 32B (16 bf16) global -> LDS, async on CDNA5 when available
__device__ __forceinline__ void stage32B(const bf16* __restrict__ g, bf16* l) {
#if HAVE_ASYNC_LDS
  __builtin_amdgcn_global_load_async_to_lds_b128((AS1 i32x4*)g,       (AS3 i32x4*)l,       0, 0);
  __builtin_amdgcn_global_load_async_to_lds_b128((AS1 i32x4*)(g + 8), (AS3 i32x4*)(l + 8), 0, 0);
#else
#pragma unroll
  for (int j = 0; j < 16; ++j) l[j] = g[j];
#endif
}

__device__ __forceinline__ void async_stage_wait() {
#if HAVE_ASYNC_LDS
  asm volatile("s_wait_asynccnt 0x0" ::: "memory");
#endif
}

__global__ void __launch_bounds__(256) k_f32_to_bf16(const float* __restrict__ in,
                                                     bf16* __restrict__ out, int n) {
  int i = blockIdx.x * blockDim.x + threadIdx.x;
  if (i < n) out[i] = (bf16)in[i];
}

// weights [4][K][Ncols] f32 -> transposed bf16 [4][Ncols][K]
__global__ void __launch_bounds__(256)
k_f32_to_bf16_tr(const float* __restrict__ in, bf16* __restrict__ out,
                 int K, int Ncols, int total) {
  int i = blockIdx.x * blockDim.x + threadIdx.x;
  if (i >= total) return;
  int per = K * Ncols;
  int r = i / per, rem = i % per;
  int n = rem / K, k = rem % K;
  out[i] = (bf16)in[(size_t)r * per + (size_t)k * Ncols + n];
}

__global__ void __launch_bounds__(256) k_zero_f32(float* __restrict__ p, int n) {
  int i = blockIdx.x * blockDim.x + threadIdx.x;
  if (i < n) p[i] = 0.f;
}

// acc[n][c] = biasA[c] + biasB[c]   (c in [0,512))
__global__ void __launch_bounds__(256) k_init_acc_bias(float* __restrict__ acc,
                                                       const float* __restrict__ bA,
                                                       const float* __restrict__ bB, int n) {
  int i = blockIdx.x * blockDim.x + threadIdx.x;
  if (i < n) { int c = i & (HH - 1); acc[i] = bA[c] + bB[c]; }
}

__global__ void __launch_bounds__(256) k_init_ms(float* __restrict__ m,
                                                 float* __restrict__ s, int n) {
  int i = blockIdx.x * blockDim.x + threadIdx.x;
  if (i < n) { m[i] = -__builtin_inff(); s[i] = 0.f; }
}

__global__ void __launch_bounds__(256) k_fix_m(float* __restrict__ m, int n) {
  int i = blockIdx.x * blockDim.x + threadIdx.x;
  if (i < n) { float v = m[i]; if (v < -3.0e38f) m[i] = 0.f; }
}

// ---------------------------------------------------------------------------
// bf16 WMMA GEMM:  C[M,N] = A[M,K] * Bt[N,K]^T   (A,Bt bf16 row-major, C f32)
// Bt is the pre-transposed weight matrix, so BOTH tiles stage as contiguous
// 32B copies (async global->LDS on CDNA5). block = 128 threads (4 waves),
// tile 64x64, K-chunk 32.
// ---------------------------------------------------------------------------
#define GTM 64
#define GTN 64
#define GTK 32

__global__ void __launch_bounds__(128)
k_gemm_bf16_wmma(const bf16* __restrict__ A, const bf16* __restrict__ Bt,
                 float* __restrict__ C, int M, int N, int K) {
  __shared__ bf16 As[GTM][GTK];   // row-major A tile
  __shared__ bf16 Bs[GTN][GTK];   // row-major Bt tile (== B columns, k contiguous)

  const int tid  = threadIdx.x;
  const int wave = tid >> 5;
  const int lane = tid & 31;
  const int m0 = blockIdx.y * GTM;
  const int n0 = blockIdx.x * GTN;
  const int r  = tid >> 1;            // 0..63 tile row handled by this thread
  const int h  = (tid & 1) * 16;      // 0 or 16: k-half handled by this thread
  const bool full = (m0 + GTM <= M);  // interior block -> no row guard needed

  f32x8 acc[4] = {};   // this wave: rows [wave*16,wave*16+16) x 64 cols

  for (int k0 = 0; k0 < K; k0 += GTK) {
    // ---- stage A: 64x32 (32B per thread) ----
    const bf16* srcA = A + (size_t)(m0 + r) * K + k0 + h;
    if (full) {
      if (k0 + GTK < K) __builtin_prefetch(srcA + GTK, 0, 1);
      stage32B(srcA, &As[r][h]);
    } else {
      if (m0 + r < M) {
#pragma unroll
        for (int j = 0; j < 16; ++j) As[r][h + j] = srcA[j];
      } else {
#pragma unroll
        for (int j = 0; j < 16; ++j) As[r][h + j] = (bf16)0.f;
      }
    }
    // ---- stage Bt: 64x32 (32B per thread), always interior ----
    stage32B(Bt + (size_t)(n0 + r) * K + k0 + h, &Bs[r][h]);

    async_stage_wait();
    __syncthreads();

    // ---- fragments + 4 WMMAs ----
    const int mrow  = wave * 16 + (lane & 15);
    const int kbase = (lane >> 4) * 8;
    bf16x16 afrag;
#pragma unroll
    for (int e = 0; e < 8; ++e) {
      afrag[e]     = As[mrow][kbase + e];
      afrag[e + 8] = As[mrow][16 + kbase + e];
    }
#pragma unroll
    for (int nt = 0; nt < 4; ++nt) {
      const int ncol = nt * 16 + (lane & 15);
      bf16x16 bfrag;
#pragma unroll
      for (int e = 0; e < 8; ++e) {
        bfrag[e]     = Bs[ncol][kbase + e];
        bfrag[e + 8] = Bs[ncol][16 + kbase + e];
      }
      acc[nt] = __builtin_amdgcn_wmma_f32_16x16x32_bf16(
          false, afrag, false, bfrag, (short)0, acc[nt], false, false);
    }
    __syncthreads();
  }

  // ---- store C (f32 16x16 layout: VGPR v -> m = (lane>=16)*8 + v) ----
  const int mbase = m0 + wave * 16 + (lane >> 4) * 8;
  const int nbase = n0 + (lane & 15);
#pragma unroll
  for (int nt = 0; nt < 4; ++nt) {
#pragma unroll
    for (int v = 0; v < 8; ++v) {
      int gm = mbase + v;
      if (gm < M) C[(size_t)gm * N + nbase + nt * 16] = acc[nt][v];
    }
  }
}

// ---------------------------------------------------------------------------
// Edge phase. One wave32 per edge; lane handles 16 of 512 channels;
// a head (128 ch) spans exactly 8 lanes.
// Edges: e < nE -> (ei[e], ei[nE+e]); e >= nE -> self loop node (e-nE).
// ---------------------------------------------------------------------------
__global__ void __launch_bounds__(256)
k_edge_logit_max(const float* __restrict__ xl, const float* __restrict__ xr,
                 const int* __restrict__ ei, int nE, int nTot,
                 const float* __restrict__ att,
                 float* __restrict__ logit, float* __restrict__ mmax) {
  int gw   = (blockIdx.x * blockDim.x + threadIdx.x) >> 5;
  int lane = threadIdx.x & 31;
  if (gw >= nTot) return;
  int s, d;
  if (gw < nE) { s = ei[gw]; d = ei[nE + gw]; } else { s = d = gw - nE; }

  const float* pl = xl + (size_t)s * HH + lane * 16;
  const float* pr = xr + (size_t)d * HH + lane * 16;
  const float* pa = att + lane * 16;
  float accv = 0.f;
#pragma unroll
  for (int j = 0; j < 16; ++j) {
    float v = pl[j] + pr[j];
    v = (v > 0.f) ? v : 0.2f * v;      // leaky_relu(0.2)
    accv += v * pa[j];
  }
  // reduce within each 8-lane head group
  accv += __shfl_xor(accv, 1, 32);
  accv += __shfl_xor(accv, 2, 32);
  accv += __shfl_xor(accv, 4, 32);
  if ((lane & 7) == 0) {
    int hh = lane >> 3;
    logit[(size_t)gw * NHEAD + hh] = accv;
    atomicMaxF(&mmax[(size_t)d * NHEAD + hh], accv);
  }
}

__global__ void __launch_bounds__(256)
k_edge_softmax(const float* __restrict__ logit, const float* __restrict__ mmax,
               const int* __restrict__ ei, int nE, int nTot,
               float* __restrict__ pexp, float* __restrict__ ssum) {
  int i = blockIdx.x * blockDim.x + threadIdx.x;
  if (i >= nTot * NHEAD) return;
  int e = i >> 2, hh = i & 3;
  int d = (e < nE) ? ei[nE + e] : (e - nE);
  float p = __expf(logit[i] - mmax[d * NHEAD + hh]);
  pexp[i] = p;
  atomicAdd(&ssum[d * NHEAD + hh], p);
}

__global__ void __launch_bounds__(256)
k_edge_scatter(const float* __restrict__ pexp, const float* __restrict__ ssum,
               const float* __restrict__ xl, const int* __restrict__ ei,
               int nE, int nTot, float* __restrict__ acc) {
  int gw   = (blockIdx.x * blockDim.x + threadIdx.x) >> 5;
  int lane = threadIdx.x & 31;
  if (gw >= nTot) return;
  int s, d;
  if (gw < nE) { s = ei[gw]; d = ei[nE + gw]; } else { s = d = gw - nE; }
  int hh = lane >> 3;
  float p  = pexp[(size_t)gw * NHEAD + hh];
  float sd = ssum[(size_t)d * NHEAD + hh];
  float alpha = p / fmaxf(sd, 1e-16f);
  const float* pl = xl + (size_t)s * HH + lane * 16;
  float* pa = acc + (size_t)d * HH + lane * 16;
#pragma unroll
  for (int j = 0; j < 16; ++j) atomicAdd(&pa[j], alpha * pl[j]);
}

// out[n][c] = tanh(sum_h acc[n][h*128+c]),  n over N, c over 128
__global__ void __launch_bounds__(256)
k_headsum_tanh(const float* __restrict__ acc, float* __restrict__ out, int n) {
  int i = blockIdx.x * blockDim.x + threadIdx.x;
  if (i >= n) return;
  int node = i >> 7, c = i & 127;
  const float* p = acc + (size_t)node * HH + c;
  out[i] = tanhf(p[0] + p[128] + p[256] + p[384]);
}

__global__ void __launch_bounds__(256)
k_pool_scatter(const float* __restrict__ x, const int* __restrict__ batch,
               float* __restrict__ pool, int n) {
  int i = blockIdx.x * blockDim.x + threadIdx.x;
  if (i >= n) return;
  int node = i >> 7, c = i & 127;
  atomicAdd(&pool[batch[node] * CH + c], x[i]);
}

// grid = B blocks, 64 threads. out layout: [logits(64) | y_i(64*64) | y_j(64*64)]
__global__ void __launch_bounds__(64)
k_final(const float* __restrict__ pool_i, const float* __restrict__ pool_j,
        const float* __restrict__ W_i, const float* __restrict__ b_i,
        const float* __restrict__ W_j, const float* __restrict__ b_j,
        const float* __restrict__ W_out, const float* __restrict__ b_out,
        float* __restrict__ out) {
  __shared__ float ti[CH], tj[CH];
  int b = blockIdx.x, t = threadIdx.x;
  ti[t]      = tanhf(pool_i[b * CH + t]);
  ti[t + 64] = tanhf(pool_i[b * CH + 64 + t]);
  tj[t]      = tanhf(pool_j[b * CH + t]);
  tj[t + 64] = tanhf(pool_j[b * CH + 64 + t]);
  __syncthreads();
  float yi = b_i[t], yj = b_j[t];
#pragma unroll 4
  for (int c = 0; c < CH; ++c) {
    yi += ti[c] * W_i[c * NINNER + t];
    yj += tj[c] * W_j[c * NINNER + t];
  }
  out[NB + b * NINNER + t] = yi;
  out[NB + NB * NINNER + b * NINNER + t] = yj;
  if (t == 0) {
    float z = b_out[0];
    for (int c = 0; c < CH; ++c) z += (ti[c] + tj[c]) * W_out[c];
    out[b] = 1.f / (1.f + __expf(-z));
  }
}

// ---------------------------------------------------------------------------
// host orchestration
// ---------------------------------------------------------------------------
extern "C" void kernel_launch(void* const* d_in, const int* in_sizes, int n_in,
                              void* d_out, int out_size, void* d_ws, size_t ws_size,
                              hipStream_t stream) {
  const float* x_i   = (const float*)d_in[0];
  const float* x_j   = (const float*)d_in[1];
  const int*   ei_ii = (const int*)d_in[2];
  const int*   ei_jj = (const int*)d_in[3];
  const int*   ei_ij = (const int*)d_in[4];
  const int*   ei_ji = (const int*)d_in[5];
  const int*   bat_i = (const int*)d_in[6];
  const int*   bat_j = (const int*)d_in[7];
  const float* Wl[2]  = {(const float*)d_in[8],  (const float*)d_in[12]};
  const float* Wr[2]  = {(const float*)d_in[9],  (const float*)d_in[13]};
  const float* att[2] = {(const float*)d_in[10], (const float*)d_in[14]};
  const float* bv[2]  = {(const float*)d_in[11], (const float*)d_in[15]};
  const float* W_out = (const float*)d_in[16];
  const float* b_out = (const float*)d_in[17];
  const float* W_i   = (const float*)d_in[18];
  const float* b_i   = (const float*)d_in[19];
  const float* W_j   = (const float*)d_in[20];
  const float* b_j   = (const float*)d_in[21];
  float* out = (float*)d_out;

  const int Nn = in_sizes[6];            // nodes per type (20000)
  const int E  = in_sizes[2] / 2;        // edges per relation (320000)
  const int ET = E + Nn;                 // + self loops
  const int F  = in_sizes[0] / Nn;       // 512

  // workspace carve-out
  char* w = (char*)d_ws;
  size_t off = 0;
  auto carve = [&](size_t bytes) -> void* {
    void* p = w + off;
    off += (bytes + 255) & ~(size_t)255;
    return p;
  };
  bf16* xb_i  = (bf16*)carve((size_t)Nn * HH * 2);
  bf16* xb_j  = (bf16*)carve((size_t)Nn * HH * 2);
  bf16* wlb0  = (bf16*)carve((size_t)4 * F  * HH * 2);   // transposed [4][HH][F]
  bf16* wrb0  = (bf16*)carve((size_t)4 * F  * HH * 2);
  bf16* wlb1  = (bf16*)carve((size_t)4 * CH * HH * 2);   // transposed [4][HH][CH]
  bf16* wrb1  = (bf16*)carve((size_t)4 * CH * HH * 2);
  float* xl    = (float*)carve((size_t)Nn * HH * 4);
  float* xr    = (float*)carve((size_t)Nn * HH * 4);
  float* acc_i = (float*)carve((size_t)Nn * HH * 4);
  float* acc_j = (float*)carve((size_t)Nn * HH * 4);
  float* logit = (float*)carve((size_t)ET * NHEAD * 4);
  float* pexp  = (float*)carve((size_t)ET * NHEAD * 4);
  float* mmax  = (float*)carve((size_t)Nn * NHEAD * 4);
  float* ssum  = (float*)carve((size_t)Nn * NHEAD * 4);
  float* xo_i  = (float*)carve((size_t)Nn * CH * 4);
  float* xo_j  = (float*)carve((size_t)Nn * CH * 4);
  float* pool_i = (float*)carve((size_t)NB * CH * 4);
  float* pool_j = (float*)carve((size_t)NB * CH * 4);

  bf16* Wlb[2] = {wlb0, wlb1};
  bf16* Wrb[2] = {wrb0, wrb1};
  const int Kin[2] = {F, CH};

  // convert all weights to bf16, transposed to [4][HH][K] for contiguous staging
  {
    int n0 = 4 * F * HH, n1 = 4 * CH * HH;
    k_f32_to_bf16_tr<<<CEILDIV(n0, 256), 256, 0, stream>>>(Wl[0], wlb0, F,  HH, n0);
    k_f32_to_bf16_tr<<<CEILDIV(n0, 256), 256, 0, stream>>>(Wr[0], wrb0, F,  HH, n0);
    k_f32_to_bf16_tr<<<CEILDIV(n1, 256), 256, 0, stream>>>(Wl[1], wlb1, CH, HH, n1);
    k_f32_to_bf16_tr<<<CEILDIV(n1, 256), 256, 0, stream>>>(Wr[1], wrb1, CH, HH, n1);
  }

  for (int l = 0; l < 2; ++l) {
    const int K = Kin[l];
    // activations -> bf16
    if (l == 0) {
      int n = Nn * F;
      k_f32_to_bf16<<<CEILDIV(n, 256), 256, 0, stream>>>(x_i, xb_i, n);
      k_f32_to_bf16<<<CEILDIV(n, 256), 256, 0, stream>>>(x_j, xb_j, n);
    } else {
      int n = Nn * CH;
      k_f32_to_bf16<<<CEILDIV(n, 256), 256, 0, stream>>>(xo_i, xb_i, n);
      k_f32_to_bf16<<<CEILDIV(n, 256), 256, 0, stream>>>(xo_j, xb_j, n);
    }
    // acc = bias sums (HeteroConv aggr='sum' of two relations per dst type)
    k_init_acc_bias<<<CEILDIV(Nn * HH, 256), 256, 0, stream>>>(acc_i, bv[l] + 0 * HH, bv[l] + 3 * HH, Nn * HH);
    k_init_acc_bias<<<CEILDIV(Nn * HH, 256), 256, 0, stream>>>(acc_j, bv[l] + 1 * HH, bv[l] + 2 * HH, Nn * HH);

    struct RelDesc { int wi; const bf16* xs; const bf16* xd; const int* ei; float* acc; };
    const RelDesc rel[4] = {
        {0, xb_i, xb_i, ei_ii, acc_i},   // i -> i
        {3, xb_j, xb_i, ei_ji, acc_i},   // j -> i
        {1, xb_j, xb_j, ei_jj, acc_j},   // j -> j
        {2, xb_i, xb_j, ei_ij, acc_j},   // i -> j
    };

    dim3 gg(HH / GTN, CEILDIV(Nn, GTM));
    for (int r = 0; r < 4; ++r) {
      const RelDesc& R = rel[r];
      k_gemm_bf16_wmma<<<gg, 128, 0, stream>>>(R.xs, Wlb[l] + (size_t)R.wi * K * HH, xl, Nn, HH, K);
      k_gemm_bf16_wmma<<<gg, 128, 0, stream>>>(R.xd, Wrb[l] + (size_t)R.wi * K * HH, xr, Nn, HH, K);
      k_init_ms<<<CEILDIV(Nn * NHEAD, 256), 256, 0, stream>>>(mmax, ssum, Nn * NHEAD);
      k_edge_logit_max<<<CEILDIV(ET, 8), 256, 0, stream>>>(xl, xr, R.ei, E, ET,
                                                           att[l] + (size_t)R.wi * HH, logit, mmax);
      k_fix_m<<<CEILDIV(Nn * NHEAD, 256), 256, 0, stream>>>(mmax, Nn * NHEAD);
      k_edge_softmax<<<CEILDIV(ET * NHEAD, 256), 256, 0, stream>>>(logit, mmax, R.ei, E, ET, pexp, ssum);
      k_edge_scatter<<<CEILDIV(ET, 8), 256, 0, stream>>>(pexp, ssum, xl, R.ei, E, ET, R.acc);
    }
    k_headsum_tanh<<<CEILDIV(Nn * CH, 256), 256, 0, stream>>>(acc_i, xo_i, Nn * CH);
    k_headsum_tanh<<<CEILDIV(Nn * CH, 256), 256, 0, stream>>>(acc_j, xo_j, Nn * CH);
  }

  // pooling + heads
  k_zero_f32<<<CEILDIV(NB * CH, 256), 256, 0, stream>>>(pool_i, NB * CH);
  k_zero_f32<<<CEILDIV(NB * CH, 256), 256, 0, stream>>>(pool_j, NB * CH);
  k_pool_scatter<<<CEILDIV(Nn * CH, 256), 256, 0, stream>>>(xo_i, bat_i, pool_i, Nn * CH);
  k_pool_scatter<<<CEILDIV(Nn * CH, 256), 256, 0, stream>>>(xo_j, bat_j, pool_j, Nn * CH);
  k_final<<<NB, 64, 0, stream>>>(pool_i, pool_j, W_i, b_i, W_j, b_j, W_out, b_out, out);
}
